// ONIMemoryHub_69741678953015
// MI455X (gfx1250) — compile-verified, matrix operand order checked by reading, not run on hardware
//
#include <hip/hip_runtime.h>
#include <hip/hip_bf16.h>

// ---------------------------------------------------------------------------
// Types for CDNA5 WMMA (wave32, 16x16x32 bf16 -> f32)
// ---------------------------------------------------------------------------
typedef __attribute__((ext_vector_type(16))) __bf16         v16bf;
typedef __attribute__((ext_vector_type(16))) unsigned short v16u;
typedef __attribute__((ext_vector_type(8)))  unsigned short v8u;
typedef __attribute__((ext_vector_type(8)))  float          v8f;
typedef __attribute__((ext_vector_type(4)))  int            v4i;

#if __has_builtin(__builtin_amdgcn_global_load_async_to_lds_b128) && \
    __has_builtin(__builtin_amdgcn_s_wait_asynccnt)
#define ONI_USE_ASYNC 1
typedef __attribute__((address_space(1))) v4i* gptr_v4i;
typedef __attribute__((address_space(3))) v4i* lptr_v4i;
#else
#define ONI_USE_ASYNC 0
#endif

__device__ __forceinline__ unsigned short f2bf(float f) {
  unsigned u = __float_as_uint(f);
  unsigned r = u + 0x7FFFu + ((u >> 16) & 1u);   // round-to-nearest-even
  return (unsigned short)(r >> 16);
}

// ---------------------------------------------------------------------------
// Tiled bf16 WMMA GEMM:  C[M,N] (f32) = A[M,K] (bf16) @ Bt[N,K]^T (bf16)
// B is ALWAYS row-major [N,K]: weights pre-transposed once; similarity GEMMs
// use key matrices that are naturally [N,K].
// Block: 256 threads (8 waves), 128x128 tile, K-step 32.
// Each wave: 64x32 patch = 4x2 tiles of v_wmma_f32_16x16x32_bf16.
// Double-buffered LDS; staging via GLOBAL_LOAD_ASYNC_TO_LDS_B128 pipelined
// against WMMA compute (drain to 4 outstanding = previous tile resident).
// ---------------------------------------------------------------------------
#define TM 128
#define TN 128
#define TK 32
#define LDST 48   // LDS row stride in bf16 elems (96B, 16B-aligned)

__global__ __launch_bounds__(256) void k_gemm_bf16(
    const unsigned short* __restrict__ A,
    const unsigned short* __restrict__ Bt,
    float* __restrict__ C,
    int Mdim, int Ndim, int Kdim)
{
  __shared__ __align__(16) unsigned short lA[2][TM * LDST];
  __shared__ __align__(16) unsigned short lB[2][TN * LDST];

  const int tid  = threadIdx.x;
  const int lane = tid & 31;
  const int wave = tid >> 5;
  const int wm   = wave >> 2;     // 0..1  (M direction)
  const int wn   = wave & 3;      // 0..3  (N direction)
  const int half = lane >> 4;     // 0/1   (K-half select per ISA layout)
  const int l16  = lane & 15;
  const int kh   = half * 8;

  const int mBase = blockIdx.y * TM;
  const int nBase = blockIdx.x * TN;

  // staging indices for this thread (2 chunks of 8 bf16 per matrix per tile)
  const int srow0 = tid >> 2,         sseg0 = (tid & 3) * 8;
  const int srow1 = (tid + 256) >> 2, sseg1 = ((tid + 256) & 3) * 8;

  auto issue = [&](int buf, int kBase) {
    const unsigned short* ga0 = A  + (size_t)(mBase + srow0) * Kdim + kBase + sseg0;
    const unsigned short* gb0 = Bt + (size_t)(nBase + srow0) * Kdim + kBase + sseg0;
    const unsigned short* ga1 = A  + (size_t)(mBase + srow1) * Kdim + kBase + sseg1;
    const unsigned short* gb1 = Bt + (size_t)(nBase + srow1) * Kdim + kBase + sseg1;
    unsigned short* la0 = lA[buf] + srow0 * LDST + sseg0;
    unsigned short* lb0 = lB[buf] + srow0 * LDST + sseg0;
    unsigned short* la1 = lA[buf] + srow1 * LDST + sseg1;
    unsigned short* lb1 = lB[buf] + srow1 * LDST + sseg1;
#if ONI_USE_ASYNC
    __builtin_amdgcn_global_load_async_to_lds_b128((gptr_v4i)ga0, (lptr_v4i)la0, 0, 0);
    __builtin_amdgcn_global_load_async_to_lds_b128((gptr_v4i)gb0, (lptr_v4i)lb0, 0, 0);
    __builtin_amdgcn_global_load_async_to_lds_b128((gptr_v4i)ga1, (lptr_v4i)la1, 0, 0);
    __builtin_amdgcn_global_load_async_to_lds_b128((gptr_v4i)gb1, (lptr_v4i)lb1, 0, 0);
#else
    *(uint4*)la0 = *(const uint4*)ga0;
    *(uint4*)lb0 = *(const uint4*)gb0;
    *(uint4*)la1 = *(const uint4*)ga1;
    *(uint4*)lb1 = *(const uint4*)gb1;
#endif
  };

  v8f acc[4][2] = {};

  const int nk = Kdim / TK;
  issue(0, 0);                        // preload tile 0

  for (int kt = 0; kt < nk; ++kt) {
    if (kt + 1 < nk) {
      issue((kt + 1) & 1, (kt + 1) * TK);   // stream next tile during compute
#if ONI_USE_ASYNC
      __builtin_amdgcn_s_wait_asynccnt(4);  // 4 newest outstanding = next tile;
                                            // tile kt is resident (in-order)
#endif
    } else {
#if ONI_USE_ASYNC
      __builtin_amdgcn_s_wait_asynccnt(0);
#endif
    }
    if (kt + 2 < nk) {                // warm L2 two tiles ahead
      __builtin_prefetch(A + (size_t)(mBase + srow0) * Kdim + (kt + 2) * TK, 0, 1);
    }
    __syncthreads();                  // all waves' tile-kt data visible

    const unsigned short* la = lA[kt & 1];
    const unsigned short* lb = lB[kt & 1];

    // ---- build fragments per ISA 16-bit A/B layout & issue 8 WMMAs ----
    union Frag { v16u v; v8u h[2]; } afr[4], bfr[2];
    #pragma unroll
    for (int sm = 0; sm < 4; ++sm) {
      int row = wm * 64 + sm * 16 + l16;
      afr[sm].h[0] = *(const v8u*)(la + row * LDST + kh);        // K = kh..kh+7
      afr[sm].h[1] = *(const v8u*)(la + row * LDST + 16 + kh);   // K = 16+kh..
    }
    #pragma unroll
    for (int sn = 0; sn < 2; ++sn) {
      int col = wn * 32 + sn * 16 + l16;
      bfr[sn].h[0] = *(const v8u*)(lb + col * LDST + kh);
      bfr[sn].h[1] = *(const v8u*)(lb + col * LDST + 16 + kh);
    }
    #pragma unroll
    for (int sm = 0; sm < 4; ++sm)
      #pragma unroll
      for (int sn = 0; sn < 2; ++sn)
        acc[sm][sn] = __builtin_amdgcn_wmma_f32_16x16x32_bf16(
            false, __builtin_bit_cast(v16bf, afr[sm].v),
            false, __builtin_bit_cast(v16bf, bfr[sn].v),
            (short)0, acc[sm][sn], false, false);
    __syncthreads();                  // done reading buf (kt&1) before reuse
  }

  // ---- epilogue: C/D layout: VGPR r, lanes0-15 -> M=r, lanes16-31 -> M=8+r
  const size_t ldc = (size_t)Ndim;
  #pragma unroll
  for (int sm = 0; sm < 4; ++sm) {
    int rowb = mBase + wm * 64 + sm * 16 + 8 * half;
    #pragma unroll
    for (int sn = 0; sn < 2; ++sn) {
      float* cp = C + (size_t)rowb * ldc + (nBase + wn * 32 + sn * 16 + l16);
      #pragma unroll
      for (int r = 0; r < 8; ++r) cp[(size_t)r * ldc] = acc[sm][sn][r];
    }
  }
}

// ---------------------------------------------------------------------------
// Elementwise f32 -> bf16 cast
// ---------------------------------------------------------------------------
__global__ void k_cast_bf16(const float* __restrict__ in,
                            unsigned short* __restrict__ out, size_t n) {
  size_t i = (size_t)blockIdx.x * blockDim.x + threadIdx.x;
  size_t stride = (size_t)gridDim.x * blockDim.x;
  for (; i < n; i += stride) out[i] = f2bf(in[i]);
}

// ---------------------------------------------------------------------------
// Tiled transpose + cast: in f32 [R,C] -> out bf16 [C,R]
// ---------------------------------------------------------------------------
__global__ __launch_bounds__(256) void k_cast_t_bf16(
    const float* __restrict__ in, unsigned short* __restrict__ out,
    int R, int C) {
  __shared__ unsigned short tile[32][33];
  int c0 = blockIdx.x * 32, r0 = blockIdx.y * 32;
  int tx = threadIdx.x & 31, ty = threadIdx.x >> 5;  // ty 0..7
  #pragma unroll
  for (int i = ty; i < 32; i += 8)
    tile[i][tx] = f2bf(in[(size_t)(r0 + i) * C + c0 + tx]);
  __syncthreads();
  #pragma unroll
  for (int i = ty; i < 32; i += 8)
    out[(size_t)(c0 + i) * R + r0 + tx] = tile[tx][i];
}

// ---------------------------------------------------------------------------
// Row L2-normalize (F.normalize semantics) f32 -> bf16; block per row
// ---------------------------------------------------------------------------
__global__ __launch_bounds__(256) void k_rownorm_bf16(
    const float* __restrict__ in, unsigned short* __restrict__ out, int cols) {
  int row = blockIdx.x;
  const float* x = in + (size_t)row * cols;
  float ss = 0.f;
  for (int c = threadIdx.x; c < cols; c += 256) { float v = x[c]; ss += v * v; }
  __shared__ float red[256];
  red[threadIdx.x] = ss; __syncthreads();
  for (int s = 128; s > 0; s >>= 1) {
    if (threadIdx.x < s) red[threadIdx.x] += red[threadIdx.x + s];
    __syncthreads();
  }
  float inv = 1.0f / fmaxf(sqrtf(red[0]), 1e-12f);
  unsigned short* o = out + (size_t)row * cols;
  for (int c = threadIdx.x; c < cols; c += 256) o[c] = f2bf(x[c] * inv);
}

// ---------------------------------------------------------------------------
// Episodic recency weights: w = exp(-|1-ts|*0.01)*(imp+1); then / (sum+1e-8)
// ---------------------------------------------------------------------------
__global__ __launch_bounds__(256) void k_epw(const float* __restrict__ imp,
                                             const float* __restrict__ ts,
                                             float* __restrict__ w,
                                             float* __restrict__ sum, int n) {
  float local = 0.f;
  for (int i = threadIdx.x; i < n; i += 256) {
    float age = 1.0f - ts[i];
    float rec = __expf(-fabsf(age) * 0.01f);
    float wi  = rec * (imp[i] + 1.0f);
    w[i] = wi; local += wi;
  }
  __shared__ float red[256];
  red[threadIdx.x] = local; __syncthreads();
  for (int s = 128; s > 0; s >>= 1) {
    if (threadIdx.x < s) red[threadIdx.x] += red[threadIdx.x + s];
    __syncthreads();
  }
  if (threadIdx.x == 0) sum[0] = red[0] + 1e-8f;
}
__global__ void k_epw_norm(float* __restrict__ w, const float* __restrict__ sum, int n) {
  int i = blockIdx.x * blockDim.x + threadIdx.x;
  if (i < n) w[i] /= sum[0];
}

// ---------------------------------------------------------------------------
// Working memory attention over S=64 slots; block per batch row
// ---------------------------------------------------------------------------
__global__ __launch_bounds__(256) void k_work_attn(
    const float* __restrict__ q, const float* __restrict__ slots,
    float* __restrict__ out, int H, int S) {
  int b = blockIdx.x;
  const float* qr = q + (size_t)b * H;
  __shared__ float part[256];
  __shared__ float attn[64];
  int s = threadIdx.x & 63;
  int stripe = threadIdx.x >> 6;       // 0..3
  int seg = H / 4;
  float acc = 0.f;
  for (int k = stripe * seg; k < (stripe + 1) * seg; ++k)
    acc += qr[k] * slots[(size_t)s * H + k];
  part[threadIdx.x] = acc; __syncthreads();
  if (threadIdx.x < 64) {
    float d = part[threadIdx.x] + part[threadIdx.x + 64] +
              part[threadIdx.x + 128] + part[threadIdx.x + 192];
    attn[threadIdx.x] = d * rsqrtf((float)H);
  }
  __syncthreads();
  if (threadIdx.x == 0) {
    float mx = attn[0];
    for (int i = 1; i < 64; ++i) mx = fmaxf(mx, attn[i]);
    float sm = 0.f;
    for (int i = 0; i < 64; ++i) { attn[i] = __expf(attn[i] - mx); sm += attn[i]; }
    float inv = 1.0f / sm;
    for (int i = 0; i < 64; ++i) attn[i] *= inv;
  }
  __syncthreads();
  for (int h = threadIdx.x; h < H; h += 256) {
    float o = 0.f;
    for (int i = 0; i < 64; ++i) o += attn[i] * slots[(size_t)i * H + h];
    out[(size_t)b * H + h] = o;
  }
}

// ---------------------------------------------------------------------------
// Router gate: softmax(silu(q@W1+b1)@W2+b2); 64 threads, block per row
// ---------------------------------------------------------------------------
__global__ __launch_bounds__(64) void k_gate(
    const float* __restrict__ q, const float* __restrict__ W1,
    const float* __restrict__ b1, const float* __restrict__ W2,
    const float* __restrict__ b2, float* __restrict__ gw, int H) {
  int b = blockIdx.x;
  const float* qr = q + (size_t)b * H;
  __shared__ float h[64];
  int j = threadIdx.x;
  float acc = b1[j];
  for (int k = 0; k < H; ++k) acc += qr[k] * W1[(size_t)k * 64 + j];
  h[j] = acc / (1.0f + __expf(-acc));   // silu
  __syncthreads();
  if (j == 0) {
    float g[3];
    for (int i = 0; i < 3; ++i) {
      float a = b2[i];
      for (int t = 0; t < 64; ++t) a += h[t] * W2[t * 3 + i];
      g[i] = a;
    }
    float mx = fmaxf(g[0], fmaxf(g[1], g[2]));
    float e0 = __expf(g[0] - mx), e1 = __expf(g[1] - mx), e2 = __expf(g[2] - mx);
    float inv = 1.0f / (e0 + e1 + e2);
    gw[(size_t)b * 3 + 0] = e0 * inv;
    gw[(size_t)b * 3 + 1] = e1 * inv;
    gw[(size_t)b * 3 + 2] = e2 * inv;
  }
}

// ---------------------------------------------------------------------------
// Top-K + softmax + gather-weighted-sum of values; block per row.
// Optional per-column weight w (episodic) applied before ranking.
// ---------------------------------------------------------------------------
template<int K>
__global__ __launch_bounds__(256) void k_topk_attend(
    const float* __restrict__ sim, const float* __restrict__ w,
    const float* __restrict__ vals, float* __restrict__ out,
    int Ncols, int H) {
  int b = blockIdx.x;
  const float* srow = sim + (size_t)b * Ncols;
  float bv[K]; int bi[K];
  #pragma unroll
  for (int i = 0; i < K; ++i) { bv[i] = -3.4e38f; bi[i] = 0; }
  for (int c = threadIdx.x; c < Ncols; c += 256) {
    float v = srow[c];
    if (w) v *= w[c];
    if (v > bv[K - 1]) {
      int p = K - 1;
      while (p > 0 && v > bv[p - 1]) { bv[p] = bv[p - 1]; bi[p] = bi[p - 1]; --p; }
      bv[p] = v; bi[p] = c;
    }
  }
  __shared__ float sv[256 * K];
  __shared__ int   si[256 * K];
  #pragma unroll
  for (int i = 0; i < K; ++i) { sv[threadIdx.x * K + i] = bv[i]; si[threadIdx.x * K + i] = bi[i]; }
  __syncthreads();
  __shared__ float attn[K];
  __shared__ int   aidx[K];
  if (threadIdx.x == 0) {
    float tv[K]; int ti[K];
    #pragma unroll
    for (int i = 0; i < K; ++i) { tv[i] = -3.4e38f; ti[i] = 0; }
    for (int c = 0; c < 256 * K; ++c) {
      float v = sv[c];
      if (v > tv[K - 1]) {
        int p = K - 1;
        while (p > 0 && v > tv[p - 1]) { tv[p] = tv[p - 1]; ti[p] = ti[p - 1]; --p; }
        tv[p] = v; ti[p] = si[c];
      }
    }
    float mx = tv[0], sm = 0.f, e[K];
    #pragma unroll
    for (int i = 0; i < K; ++i) { e[i] = __expf(tv[i] - mx); sm += e[i]; }
    #pragma unroll
    for (int i = 0; i < K; ++i) { attn[i] = e[i] / sm; aidx[i] = ti[i]; }
  }
  __syncthreads();
  for (int h = threadIdx.x; h < H; h += 256) {
    float o = 0.f;
    #pragma unroll
    for (int i = 0; i < K; ++i) o += attn[i] * vals[(size_t)aidx[i] * H + h];
    out[(size_t)b * H + h] = o;
  }
}

// ---------------------------------------------------------------------------
// Blend with gate weights -> bf16
// ---------------------------------------------------------------------------
__global__ __launch_bounds__(256) void k_blend(
    const float* __restrict__ wo, const float* __restrict__ eo,
    const float* __restrict__ so, const float* __restrict__ gw,
    unsigned short* __restrict__ out_bf, int H) {
  int b = blockIdx.x;
  float g0 = gw[(size_t)b * 3], g1 = gw[(size_t)b * 3 + 1], g2 = gw[(size_t)b * 3 + 2];
  size_t base = (size_t)b * H;
  for (int h = threadIdx.x; h < H; h += 256)
    out_bf[base + h] = f2bf(g0 * wo[base + h] + g1 * eo[base + h] + g2 * so[base + h]);
}

// ---------------------------------------------------------------------------
// In-place LayerNorm on d_out; block per row
// ---------------------------------------------------------------------------
__global__ __launch_bounds__(256) void k_layernorm(
    float* __restrict__ x, const float* __restrict__ gamma,
    const float* __restrict__ beta, int H) {
  int b = blockIdx.x;
  float* r = x + (size_t)b * H;
  __shared__ float red[256];
  float s = 0.f;
  for (int h = threadIdx.x; h < H; h += 256) s += r[h];
  red[threadIdx.x] = s; __syncthreads();
  for (int st = 128; st > 0; st >>= 1) {
    if (threadIdx.x < st) red[threadIdx.x] += red[threadIdx.x + st];
    __syncthreads();
  }
  float mu = red[0] / H;
  __syncthreads();
  float v = 0.f;
  for (int h = threadIdx.x; h < H; h += 256) { float d = r[h] - mu; v += d * d; }
  red[threadIdx.x] = v; __syncthreads();
  for (int st = 128; st > 0; st >>= 1) {
    if (threadIdx.x < st) red[threadIdx.x] += red[threadIdx.x + st];
    __syncthreads();
  }
  float inv = rsqrtf(red[0] / H + 1e-5f);
  for (int h = threadIdx.x; h < H; h += 256)
    r[h] = (r[h] - mu) * inv * gamma[h] + beta[h];
}

// ---------------------------------------------------------------------------
// Host-side orchestration
// ---------------------------------------------------------------------------
extern "C" void kernel_launch(void* const* d_in, const int* in_sizes, int n_in,
                              void* d_out, int out_size, void* d_ws, size_t ws_size,
                              hipStream_t stream) {
  const float* query       = (const float*)d_in[0];
  const float* ep_store    = (const float*)d_in[1];
  const float* ep_imp      = (const float*)d_in[2];
  const float* ep_ts       = (const float*)d_in[3];
  const float* sem_keys    = (const float*)d_in[4];
  const float* sem_values  = (const float*)d_in[5];
  const float* work_slots  = (const float*)d_in[6];
  const float* W_query     = (const float*)d_in[7];
  const float* W_ek        = (const float*)d_in[8];
  const float* W_ev        = (const float*)d_in[9];
  const float* W_eo        = (const float*)d_in[10];
  const float* W_sq        = (const float*)d_in[11];
  const float* W_sk        = (const float*)d_in[12];
  const float* W_so        = (const float*)d_in[13];
  const float* gate_W1     = (const float*)d_in[14];
  const float* gate_b1     = (const float*)d_in[15];
  const float* gate_W2     = (const float*)d_in[16];
  const float* gate_b2     = (const float*)d_in[17];
  const float* W_ro        = (const float*)d_in[18];
  const float* ln_gamma    = (const float*)d_in[19];
  const float* ln_beta     = (const float*)d_in[20];

  const int H = in_sizes[14] / 64;   // gate_W1 is [H,64]
  const int B = in_sizes[0] / H;
  const int N = in_sizes[1] / H;
  const int M = in_sizes[4] / H;
  const int S = in_sizes[6] / H;

  // bump allocator on d_ws
  char* wsp = (char*)d_ws;
  auto alloc = [&](size_t bytes) -> void* {
    void* r = wsp; wsp += (bytes + 255) & ~(size_t)255; return r;
  };

  unsigned short* query_bf = (unsigned short*)alloc((size_t)B * H * 2);
  unsigned short* ep_bf    = (unsigned short*)alloc((size_t)N * H * 2);
  unsigned short* sem_bf   = (unsigned short*)alloc((size_t)M * H * 2);
  unsigned short* WqT_bf   = (unsigned short*)alloc((size_t)H * H * 2);
  unsigned short* WekT_bf  = (unsigned short*)alloc((size_t)H * H * 2);
  unsigned short* WevT_bf  = (unsigned short*)alloc((size_t)H * H * 2);
  unsigned short* WeoT_bf  = (unsigned short*)alloc((size_t)H * H * 2);
  unsigned short* WsqT_bf  = (unsigned short*)alloc((size_t)H * H * 2);
  unsigned short* WskT_bf  = (unsigned short*)alloc((size_t)H * H * 2);
  unsigned short* WsoT_bf  = (unsigned short*)alloc((size_t)H * H * 2);
  unsigned short* WroT_bf  = (unsigned short*)alloc((size_t)H * H * 2);
  float*          q_f32    = (float*)alloc((size_t)B * H * 4);
  unsigned short* q_bf     = (unsigned short*)alloc((size_t)B * H * 2);
  unsigned short* qn_bf    = (unsigned short*)alloc((size_t)B * H * 2);
  float*          ek_f32   = (float*)alloc((size_t)N * H * 4);
  unsigned short* ekn_bf   = (unsigned short*)alloc((size_t)N * H * 2);
  float*          ev_f32   = (float*)alloc((size_t)N * H * 4);
  float*          qs_f32   = (float*)alloc((size_t)B * H * 4);
  unsigned short* qsn_bf   = (unsigned short*)alloc((size_t)B * H * 2);
  float*          ks_f32   = (float*)alloc((size_t)M * H * 4);
  unsigned short* ksn_bf   = (unsigned short*)alloc((size_t)M * H * 2);
  float*          sim_e    = (float*)alloc((size_t)B * N * 4);
  float*          sim_s    = (float*)alloc((size_t)B * M * 4);
  float*          w_out    = (float*)alloc((size_t)B * H * 4);
  float*          e_att    = (float*)alloc((size_t)B * H * 4);
  unsigned short* e_att_bf = (unsigned short*)alloc((size_t)B * H * 2);
  float*          e_out    = (float*)alloc((size_t)B * H * 4);
  float*          s_att    = (float*)alloc((size_t)B * H * 4);
  unsigned short* s_att_bf = (unsigned short*)alloc((size_t)B * H * 2);
  float*          s_out    = (float*)alloc((size_t)B * H * 4);
  float*          gw       = (float*)alloc((size_t)B * 3 * 4);
  float*          epw      = (float*)alloc((size_t)N * 4);
  float*          epw_sum  = (float*)alloc(4);
  unsigned short* blend_bf = (unsigned short*)alloc((size_t)B * H * 2);

  auto cast = [&](const float* in, unsigned short* out, size_t n) {
    size_t b = (n + 2047) / 2048; if (b > 4096) b = 4096;
    k_cast_bf16<<<(unsigned)b, 256, 0, stream>>>(in, out, n);
  };
  auto cast_t = [&](const float* in, unsigned short* out, int R, int C) {
    k_cast_t_bf16<<<dim3(C / 32, R / 32), 256, 0, stream>>>(in, out, R, C);
  };
  auto gemm = [&](const unsigned short* A, const unsigned short* Btm, float* C,
                  int Md, int Nd, int Kd) {
    k_gemm_bf16<<<dim3(Nd / TN, Md / TM), 256, 0, stream>>>(A, Btm, C, Md, Nd, Kd);
  };

  // ---- precision casts; weights are transpose-cast once (all GEMMs NT) ----
  cast(query,    query_bf, (size_t)B * H);
  cast(ep_store, ep_bf,    (size_t)N * H);
  cast(sem_keys, sem_bf,   (size_t)M * H);
  cast_t(W_query, WqT_bf,  H, H);
  cast_t(W_ek,    WekT_bf, H, H);
  cast_t(W_ev,    WevT_bf, H, H);
  cast_t(W_eo,    WeoT_bf, H, H);
  cast_t(W_sq,    WsqT_bf, H, H);
  cast_t(W_sk,    WskT_bf, H, H);
  cast_t(W_so,    WsoT_bf, H, H);
  cast_t(W_ro,    WroT_bf, H, H);

  // ---- hub query projection ----
  gemm(query_bf, WqT_bf, q_f32, B, H, H);
  cast(q_f32, q_bf, (size_t)B * H);
  k_rownorm_bf16<<<B, 256, 0, stream>>>(q_f32, qn_bf, H);

  // ---- router gate + working memory (f32 VALU, small) ----
  k_gate<<<B, 64, 0, stream>>>(q_f32, gate_W1, gate_b1, gate_W2, gate_b2, gw, H);
  k_work_attn<<<B, 256, 0, stream>>>(q_f32, work_slots, w_out, H, S);

  // ---- episodic memory ----
  gemm(ep_bf, WekT_bf, ek_f32, N, H, H);
  k_rownorm_bf16<<<N, 256, 0, stream>>>(ek_f32, ekn_bf, H);
  gemm(ep_bf, WevT_bf, ev_f32, N, H, H);
  k_epw<<<1, 256, 0, stream>>>(ep_imp, ep_ts, epw, epw_sum, N);
  k_epw_norm<<<(N + 255) / 256, 256, 0, stream>>>(epw, epw_sum, N);
  gemm(qn_bf, ekn_bf, sim_e, B, N, H);
  k_topk_attend<8><<<B, 256, 0, stream>>>(sim_e, epw, ev_f32, e_att, N, H);
  cast(e_att, e_att_bf, (size_t)B * H);
  gemm(e_att_bf, WeoT_bf, e_out, B, H, H);

  // ---- semantic memory ----
  gemm(q_bf, WsqT_bf, qs_f32, B, H, H);
  k_rownorm_bf16<<<B, 256, 0, stream>>>(qs_f32, qsn_bf, H);
  gemm(sem_bf, WskT_bf, ks_f32, M, H, H);
  k_rownorm_bf16<<<M, 256, 0, stream>>>(ks_f32, ksn_bf, H);
  gemm(qsn_bf, ksn_bf, sim_s, B, M, H);
  k_topk_attend<4><<<B, 256, 0, stream>>>(sim_s, nullptr, sem_values, s_att, M, H);
  cast(s_att, s_att_bf, (size_t)B * H);
  gemm(s_att_bf, WsoT_bf, s_out, B, H, H);

  // ---- blend, readout projection, layernorm ----
  k_blend<<<B, 256, 0, stream>>>(w_out, e_out, s_out, gw, blend_bf, H);
  gemm(blend_bf, WroT_bf, (float*)d_out, B, H, H);
  k_layernorm<<<B, 256, 0, stream>>>((float*)d_out, ln_gamma, ln_beta, H);
}
